// MultiHeadAttentionBlock_71957882077850
// MI455X (gfx1250) — matrix operand order repaired and checked by reading, not compiled, not run on hardware
//
#include <hip/hip_runtime.h>
#include <hip/hip_bf16.h>

// ---- problem constants (match reference) ----
#define Bn  2
#define Sn  2048
#define Dn  1024
#define Hn  16
#define DKn 64

typedef __attribute__((ext_vector_type(16))) __bf16 v16bf;
typedef __attribute__((ext_vector_type(8)))  __bf16 v8bf;
typedef __attribute__((ext_vector_type(4)))  __bf16 v4bf;
typedef __attribute__((ext_vector_type(8)))  float  v8f;
typedef __attribute__((ext_vector_type(4)))  float  f32x4;
typedef __attribute__((ext_vector_type(4)))  unsigned int u32x4;

// D = A(16x32 bf16) * B(32x16 bf16) + C(16x16 f32)
__device__ __forceinline__ v8f wmma_acc(v16bf a, v16bf b, v8f c) {
  return __builtin_amdgcn_wmma_f32_16x16x32_bf16(
      /*neg_a=*/false, a, /*neg_b=*/false, b,
      /*c_mod=*/(short)0, c, /*reuse_a=*/false, /*reuse_b=*/false);
}

// Load a 16-bf16-per-lane fragment as two b128 loads.
// A fragments: second run is +16 elements; B fragments: second run is +16 too.
__device__ __forceinline__ v16bf load_frag16(const __bf16* p, int off2) {
  v16bf a;
  ((u32x4*)&a)[0] = *(const u32x4*)p;
  ((u32x4*)&a)[1] = *(const u32x4*)(p + off2);
  return a;
}

// A fragment sourced from fp32 activations: 4x b128 f32 loads + cvt to bf16.
__device__ __forceinline__ v16bf load_a_f32(const float* p) {
  f32x4 x0 = *(const f32x4*)(p);
  f32x4 x1 = *(const f32x4*)(p + 4);
  f32x4 x2 = *(const f32x4*)(p + 16);
  f32x4 x3 = *(const f32x4*)(p + 20);
  v16bf a;
#pragma unroll
  for (int t = 0; t < 4; ++t) {
    a[t]      = (__bf16)x0[t];
    a[4 + t]  = (__bf16)x1[t];
    a[8 + t]  = (__bf16)x2[t];
    a[12 + t] = (__bf16)x3[t];
  }
  return a;
}

// fp32 -> bf16 bulk convert (weights), vectorized x4
__global__ __launch_bounds__(256)
void cvt_f32_bf16(const float* __restrict__ in, __bf16* __restrict__ out, int n4) {
  int i = blockIdx.x * blockDim.x + threadIdx.x;
  if (i < n4) {
    f32x4 x = ((const f32x4*)in)[i];
    v4bf y;
#pragma unroll
    for (int t = 0; t < 4; ++t) y[t] = (__bf16)x[t];
    ((v4bf*)out)[i] = y;
  }
}

// ----------------------------------------------------------------------------
// Projection GEMM: Y[r, e] = sum_d X[r, d] * W[e, d]   (r = b*S+s, 4096x1024x1024)
// Block: 8 waves (32x8), arranged 2(M) x 4(N). Wave tile = 32(M) x 64(N)
// (2 A-frags x 4 B-frags -> 8 WMMAs per k-step, ~11 FLOP/B from L2).
// Block tile = 64(M) x 256(N).
// AMODE: 0 = X is fp32 (convert in-register), 1 = X is bf16.
// SMODE: 0 = bf16 out in per-head [b,h,s,dk]   (Q,K)
//        1 = bf16 out transposed  [b,h,dk,s]   (V; packed b128 stores)
//        2 = fp32 out row-major   [b,s,e]      (final output)
// ----------------------------------------------------------------------------
template<int AMODE, int SMODE>
__global__ __launch_bounds__(256)
void proj_gemm(const void* __restrict__ Xv, const __bf16* __restrict__ W,
               void* __restrict__ Yv) {
  const int lane = threadIdx.x;
  const int l16 = lane & 15, hf = lane >> 4;
  const int wid = threadIdx.y;
  const int mBase = blockIdx.x * 64 + (wid & 1) * 32;
  const int nBase = blockIdx.y * 256 + (wid >> 1) * 64;

  const float*  Xf = (const float*)Xv;
  const __bf16* Xb = (const __bf16*)Xv;

  v8f acc[2][4] = {};

  for (int k0 = 0; k0 < Dn; k0 += 32) {
    v16bf a[2];
#pragma unroll
    for (int mt = 0; mt < 2; ++mt) {
      const int aRow = mBase + mt * 16 + l16;
      if (AMODE == 0) a[mt] = load_a_f32(Xf + (size_t)aRow * Dn + k0 + hf * 8);
      else            a[mt] = load_frag16(Xb + (size_t)aRow * Dn + k0 + hf * 8, 16);
    }
#pragma unroll
    for (int nt = 0; nt < 4; ++nt) {
      // B column e = row e of W (contiguous in d): two b128 per lane
      const __bf16* wp = W + (size_t)(nBase + nt * 16 + l16) * Dn + k0 + hf * 8;
      v16bf bW = load_frag16(wp, 16);
#pragma unroll
      for (int mt = 0; mt < 2; ++mt)
        acc[mt][nt] = wmma_acc(a[mt], bW, acc[mt][nt]);
    }
  }

#pragma unroll
  for (int mt = 0; mt < 2; ++mt) {
    const int r0 = mBase + mt * 16 + hf * 8;   // 8 consecutive rows per lane
    const int bb = r0 >> 11;                   // r / S
    const int s0 = r0 & (Sn - 1);

    if (SMODE == 0) {
      __bf16* Y = (__bf16*)Yv;
#pragma unroll
      for (int nt = 0; nt < 4; ++nt) {
        int e = nBase + nt * 16 + l16;
        __bf16* yp = Y + (((size_t)(bb * Hn + (e >> 6))) * Sn + s0) * DKn + (e & 63);
#pragma unroll
        for (int i = 0; i < 8; ++i) yp[i * DKn] = (__bf16)acc[mt][nt][i];
      }
    } else if (SMODE == 1) {
      __bf16* Y = (__bf16*)Yv;
#pragma unroll
      for (int nt = 0; nt < 4; ++nt) {
        int e = nBase + nt * 16 + l16;
        v8bf pk;
#pragma unroll
        for (int i = 0; i < 8; ++i) pk[i] = (__bf16)acc[mt][nt][i];
        *(v8bf*)(Y + (((size_t)(bb * Hn + (e >> 6))) * DKn + (e & 63)) * Sn + s0) = pk;
      }
    } else {
      float* Y = (float*)Yv;
#pragma unroll
      for (int nt = 0; nt < 4; ++nt) {
        int e = nBase + nt * 16 + l16;
#pragma unroll
        for (int i = 0; i < 8; ++i) Y[(size_t)(r0 + i) * Dn + e] = acc[mt][nt][i];
      }
    }
  }
}

// ----------------------------------------------------------------------------
// Attention with softmax over the HEAD axis.
// Workgroup = 16 waves (one per head), one 16-query tile per workgroup.
// Per 32-key tile: score WMMAs -> raw scores to LDS [q*32+k][17] ->
// each of the 512 threads reduces 16 heads for one (q,k) -> (max, 1/sum) to LDS
// -> each wave forms bf16 P in its private LDS region (stride-40 rows, bank-
// conflict-free b128 transpose readback, same-wave DS ops are in-order) ->
// 4 P@V WMMAs against transposed V.
// ----------------------------------------------------------------------------
__global__ __launch_bounds__(512)
void attn_headsoftmax(const __bf16* __restrict__ Q, const __bf16* __restrict__ K,
                      const __bf16* __restrict__ Vt, __bf16* __restrict__ X) {
  __shared__ __align__(16) float  sc[512 * 17];      // 34816 B
  __shared__ __align__(16) float2 stats[512];        //  4096 B
  __shared__ __align__(16) __bf16 pbuf[16 * 16 * 40];// 20480 B

  const int lane = threadIdx.x;
  const int h    = threadIdx.y;
  const int l16  = lane & 15, hf = lane >> 4;
  const int tid  = h * 32 + lane;
  const int b    = blockIdx.y;
  const int q0   = blockIdx.x * 16;

  const __bf16* Qb = Q  + ((size_t)(b * Hn + h) * Sn + q0) * DKn;
  const __bf16* Kb = K  + (size_t)(b * Hn + h) * Sn * DKn;
  const __bf16* Vb = Vt + (size_t)(b * Hn + h) * DKn * Sn;

  // Q tile 16x64 as two 16x32 A fragments, kept in registers for the whole pass
  v16bf aQ[2];
#pragma unroll
  for (int j = 0; j < 2; ++j)
    aQ[j] = load_frag16(Qb + (size_t)l16 * DKn + j * 32 + hf * 8, 16);

  v8f o[4] = {};   // O tile 16q x 64dk

  for (int kt = 0; kt < Sn; kt += 32) {
    // ---- S = Q K^T (this head), 16q x 32k ----
    v8f c[2] = {};
#pragma unroll
    for (int kc = 0; kc < 2; ++kc)
#pragma unroll
      for (int j = 0; j < 2; ++j) {
        v16bf bK = load_frag16(Kb + (size_t)(kt + kc * 16 + l16) * DKn + j * 32 + hf * 8, 16);
        c[kc] = wmma_acc(aQ[j], bK, c[kc]);
      }

    // ---- publish raw scores: sc[(q*32+k)*17 + h] (pad 17 -> conflict-free) ----
#pragma unroll
    for (int kc = 0; kc < 2; ++kc)
#pragma unroll
      for (int i = 0; i < 8; ++i) {
        int e = (hf * 8 + i) * 32 + kc * 16 + l16;
        sc[e * 17 + h] = c[kc][i];
      }
    __syncthreads();

    // ---- head-axis softmax stats: thread `tid` owns element (q,k)=tid ----
    {
      float v[16];
      float m = -3.0e38f;
#pragma unroll
      for (int j = 0; j < 16; ++j) { v[j] = sc[tid * 17 + j]; m = fmaxf(m, v[j]); }
      float d = 0.f;
#pragma unroll
      for (int j = 0; j < 16; ++j) d += __expf((v[j] - m) * 0.125f); // 1/sqrt(64) in exp
      stats[tid] = make_float2(m, __frcp_rn(d));
    }
    __syncthreads();

    // ---- P for this head -> bf16 in private LDS region (row stride 40) ----
#pragma unroll
    for (int kc = 0; kc < 2; ++kc)
#pragma unroll
      for (int i = 0; i < 8; ++i) {
        int q = hf * 8 + i, kk = kc * 16 + l16;
        float2 st = stats[q * 32 + kk];
        float p = __expf((c[kc][i] - st.x) * 0.125f) * st.y;
        pbuf[h * 640 + q * 40 + kk] = (__bf16)p;
      }
    // transpose readback as one 16x32 A fragment (same-wave DS: in-order)
    v16bf aP = load_frag16(pbuf + h * 640 + l16 * 40 + hf * 8, 16);

    // ---- O += P @ V  (V transposed: contiguous b128 B fragments) ----
#pragma unroll
    for (int nt = 0; nt < 4; ++nt) {
      v16bf bV = load_frag16(Vb + (size_t)(nt * 16 + l16) * Sn + kt + hf * 8, 16);
      o[nt] = wmma_acc(aP, bV, o[nt]);
    }
  }

  // x[b, q, h*64+dk] in bf16 feeds the output projection
#pragma unroll
  for (int nt = 0; nt < 4; ++nt) {
    int e = h * 64 + nt * 16 + l16;
#pragma unroll
    for (int i = 0; i < 8; ++i)
      X[((size_t)b * Sn + q0 + hf * 8 + i) * Dn + e] = (__bf16)o[nt][i];
  }
}

// ----------------------------------------------------------------------------
extern "C" void kernel_launch(void* const* d_in, const int* in_sizes, int n_in,
                              void* d_out, int out_size, void* d_ws, size_t ws_size,
                              hipStream_t stream) {
  (void)in_sizes; (void)n_in; (void)out_size; (void)ws_size;

  const float* qi = (const float*)d_in[0];
  const float* ki = (const float*)d_in[1];
  const float* vi = (const float*)d_in[2];
  // d_in[3] = mask: the reference discards masked_fill's result -> no-op
  const float* Wq = (const float*)d_in[4];
  const float* Wk = (const float*)d_in[5];
  const float* Wv = (const float*)d_in[6];
  const float* Wo = (const float*)d_in[7];

  // workspace layout (40 MB total)
  __bf16* p   = (__bf16*)d_ws;
  __bf16* Wqb = p; p += (size_t)Dn * Dn;
  __bf16* Wkb = p; p += (size_t)Dn * Dn;
  __bf16* Wvb = p; p += (size_t)Dn * Dn;
  __bf16* Wob = p; p += (size_t)Dn * Dn;
  __bf16* Qb  = p; p += (size_t)Bn * Sn * Dn;   // [B,H,S,DK]
  __bf16* Kb  = p; p += (size_t)Bn * Sn * Dn;   // [B,H,S,DK]
  __bf16* Vtb = p; p += (size_t)Bn * Sn * Dn;   // [B,H,DK,S]
  __bf16* Xat = p; p += (size_t)Bn * Sn * Dn;   // [B,S,D]

  const int n4 = Dn * Dn / 4;
  dim3 cb(256), cg((n4 + 255) / 256);
  cvt_f32_bf16<<<cg, cb, 0, stream>>>(Wq, Wqb, n4);
  cvt_f32_bf16<<<cg, cb, 0, stream>>>(Wk, Wkb, n4);
  cvt_f32_bf16<<<cg, cb, 0, stream>>>(Wv, Wvb, n4);
  cvt_f32_bf16<<<cg, cb, 0, stream>>>(Wo, Wob, n4);

  dim3 gb(32, 8);
  dim3 gg(Bn * Sn / 64, Dn / 256);
  proj_gemm<0, 0><<<gg, gb, 0, stream>>>(qi, Wqb, Qb);
  proj_gemm<0, 0><<<gg, gb, 0, stream>>>(ki, Wkb, Kb);
  proj_gemm<0, 1><<<gg, gb, 0, stream>>>(vi, Wvb, Vtb);

  attn_headsoftmax<<<dim3(Sn / 16, Bn), dim3(32, 16), 0, stream>>>(Qb, Kb, Vtb, Xat);

  proj_gemm<1, 2><<<gg, gb, 0, stream>>>(Xat, Wob, (float*)d_out);
}
